// TopoReg_51153060495999
// MI455X (gfx1250) — matrix operand order repaired and checked by reading, not compiled, not run on hardware
//
#include <hip/hip_runtime.h>

// ---------------------------------------------------------------------------
// kNN topological-regularizer for MI455X (gfx1250, wave32, WMMA).
//
// dist2(i,j) = x2[i] + x2[j] - 2 * <E[i], E[j]>   (Gram via bf16 split WMMA)
//   E ~ hi + lo (two bf16), <x,y> ~ hi.hi + hi.lo + lo.hi  (3 WMMAs / K-slab)
// 256 blocks x 8 waves. Wave (g = w&3, cq = w>>2) owns rows [g*16,+16) of the
// block and streams column tiles of half cq. The 16x128 column panel (hi+lo)
// is double-buffered in LDS via GLOBAL_LOAD_ASYNC_TO_LDS_B128 (ASYNCcnt),
// staged cooperatively by the 4 waves of each half while WMMAs run on the
// other buffer. B fragments (the wave's own 16 rows) live in VGPRs for the
// whole kernel. All 16 A-fragment ds_loads of a tile are hoisted ahead of the
// 12-WMMA chain so one dscnt wait covers them. Per-lane running top-5 with an
// execz-guarded sorted insert; final merge = bitonic-split min-pairing +
// 9-CE sort network + one LDS cross-wave merge step.
// ---------------------------------------------------------------------------

#define NPTS   16384
#define DIM    128
#define TILE   16
#define NTILES (NPTS / TILE)            // 1024
#define WPB    8                        // waves per block
#define NHALF  2                        // column-range split
#define TPW    (NTILES / NHALF)         // 512 tiles per wave
#define ROWS_PER_BLOCK 64               // 4 row groups * 16
#define NBLK   (NPTS / ROWS_PER_BLOCK)  // 256
#define BIGV   1e30f

typedef __attribute__((ext_vector_type(16))) __bf16 v16bf;
typedef __attribute__((ext_vector_type(8)))  float  v8f;

struct Chunk32 { uint4 a, b; };   // 32 bytes -> bit_cast to v16bf

__device__ __forceinline__ unsigned short f32_to_bf16_rne(float f) {
    unsigned u = __float_as_uint(f);
    unsigned r = u + 0x7FFFu + ((u >> 16) & 1u);
    return (unsigned short)(r >> 16);
}

// --- gfx1250 async global->LDS staging (ASYNCcnt path), with safe fallback --
#if __has_builtin(__builtin_amdgcn_global_load_async_to_lds_b128)
#define HAS_ASYNC_LDS 1
#else
#define HAS_ASYNC_LDS 0
#endif

typedef int i32x4v __attribute__((vector_size(16)));
typedef __attribute__((address_space(1))) i32x4v as1_i32x4;  // global
typedef __attribute__((address_space(3))) i32x4v as3_i32x4;  // LDS

__device__ __forceinline__ void stage16(const unsigned short* g,
                                        unsigned short* l) {
#if HAS_ASYNC_LDS
    // (global int4* src, lds int4* dst, imm offset, imm cpol)
    __builtin_amdgcn_global_load_async_to_lds_b128(
        (as1_i32x4*)(unsigned long long)g,
        (as3_i32x4*)(unsigned)(unsigned long long)l,   // low 32b = LDS offset
        0, 0);
#else
    *(uint4*)l = *(const uint4*)g;     // load + ds_store fallback
#endif
}

__device__ __forceinline__ void stage_wait() {
#if HAS_ASYNC_LDS
#if __has_builtin(__builtin_amdgcn_s_wait_asynccnt)
    __builtin_amdgcn_s_wait_asynccnt(0);
#else
    asm volatile("s_wait_asynccnt 0x0" ::: "memory");
#endif
#endif
}

__device__ __forceinline__ void ce(float& a, float& b) {  // compare-exchange
    float t = fminf(a, b); b = fmaxf(a, b); a = t;
}

// ---------------------------------------------------------------------------
// Phase 1: fp32 -> (hi, lo) bf16 split + per-row squared norms (true fp32).
// ---------------------------------------------------------------------------
__global__ void __launch_bounds__(DIM) prep_kernel(
        const float* __restrict__ emb,
        unsigned short* __restrict__ ehi,
        unsigned short* __restrict__ elo,
        float* __restrict__ x2) {
    const int row = blockIdx.x;
    const int t   = threadIdx.x;                 // 0..127
    float v = emb[(size_t)row * DIM + t];
    unsigned short h = f32_to_bf16_rne(v);
    float hf = __uint_as_float(((unsigned)h) << 16);
    unsigned short l = f32_to_bf16_rne(v - hf);
    ehi[(size_t)row * DIM + t] = h;
    elo[(size_t)row * DIM + t] = l;

    float s = v * v;
    #pragma unroll
    for (int off = 16; off > 0; off >>= 1) s += __shfl_xor(s, off, 32);
    __shared__ float sred[DIM / 32];
    if ((t & 31) == 0) sred[t >> 5] = s;
    __syncthreads();
    if (t == 0) x2[row] = sred[0] + sred[1] + sred[2] + sred[3];
}

// ---------------------------------------------------------------------------
// Phase 2: streaming tiled Gram + per-row top-5 + penalty partial sums.
// ---------------------------------------------------------------------------
__global__ void __launch_bounds__(WPB * 32) knn_kernel(
        const unsigned short* __restrict__ ehi16,
        const unsigned short* __restrict__ elo16,
        const float* __restrict__ x2,
        float* __restrict__ partials) {
    const int lane = threadIdx.x & 31;
    const int wave = threadIdx.x >> 5;
    const int r    = lane & 15;         // output row within wave's 16-row block
    const int h    = lane >> 4;         // half-lane select
    const int g    = wave & 3;          // row group within block
    const int cq   = wave >> 2;         // column half (0/1)
    const int ht   = threadIdx.x & 127; // thread id within half-group (4 waves)
    const int rowBase = blockIdx.x * ROWS_PER_BLOCK + g * TILE;

    // [half][buf][hi/lo][row][k] : 2*2*2*16*128*2B = 32 KB
    __shared__ __attribute__((aligned(16)))
        unsigned short panel[NHALF][2][2][TILE][DIM];
    __shared__ float smem5[WPB][TILE][5];
    __shared__ float bsum[4];

    const __bf16* ehi = (const __bf16*)ehi16;
    const __bf16* elo = (const __bf16*)elo16;

    // Resident B fragments (row block, K=128 as four K=32 slabs).
    // B layout (32x16 bf16): lane L holds K = 16*(L>>4) .. +15 of column L&15.
    v16bf Bhi[4], Blo[4];
    {
        const __bf16* bh = ehi + (size_t)(rowBase + r) * DIM + h * 16;
        const __bf16* bl = elo + (size_t)(rowBase + r) * DIM + h * 16;
        #pragma unroll
        for (int k = 0; k < 4; ++k) {
            Bhi[k] = *(const v16bf*)(bh + k * 32);
            Blo[k] = *(const v16bf*)(bl + k * 32);
        }
    }
    const float x2row = x2[rowBase + r];

    // Cooperative panel stage: 4 waves copy 8 KB (hi+lo) = 512 x 16B chunks.
    auto stage_tile = [&](int tileIdx, int buf) {
        const size_t colOff = (size_t)tileIdx * TILE * DIM;
        #pragma unroll
        for (int j = 0; j < 4; ++j) {
            const int k  = ht + 128 * j;        // 0..511
            const int hl = k >> 8;              // 0 = hi, 1 = lo
            const int e  = (k & 255) * 8;       // bf16 element offset in panel
            const unsigned short* gsrc = (hl ? elo16 : ehi16) + colOff + e;
            stage16(gsrc, &panel[cq][buf][hl][0][0] + e);
        }
    };

    // Running 5 smallest of (x2[col] - 2*dot), sorted ascending m0..m4.
    float m0 = BIGV, m1 = BIGV, m2 = BIGV, m3 = BIGV, m4 = BIGV;

    const int tile0 = cq * TPW;
    stage_tile(tile0, 0);
    stage_wait();
    __syncthreads();

    for (int i = 0; i < TPW; ++i) {
        const int buf = i & 1;
        if (i + 1 < TPW) stage_tile(tile0 + i + 1, (i + 1) & 1);  // prefetch

        const int colBase = (tile0 + i) * TILE;
        // A layout (16x32 bf16): lane L holds row (L&15), K
        // [k0 + 16h, +16)  -- read from LDS panel.
        // Hoist ALL fragment loads so one dscnt wait covers the WMMA chain.
        const unsigned short* ahp = &panel[cq][buf][0][r][h * 16];
        const unsigned short* alp = &panel[cq][buf][1][r][h * 16];
        Chunk32 cH[4], cL[4];
        #pragma unroll
        for (int k = 0; k < 4; ++k) {
            cH[k].a = *(const uint4*)(ahp + k * 32);
            cH[k].b = *(const uint4*)(ahp + k * 32 + 8);
            cL[k].a = *(const uint4*)(alp + k * 32);
            cL[k].b = *(const uint4*)(alp + k * 32 + 8);
        }
        v8f acc = {};
        #pragma unroll
        for (int k = 0; k < 4; ++k) {
            v16bf Ah = __builtin_bit_cast(v16bf, cH[k]);
            v16bf Al = __builtin_bit_cast(v16bf, cL[k]);
            acc = __builtin_amdgcn_wmma_f32_16x16x32_bf16(false, Ah, false, Bhi[k], (short)0, acc, false, false);
            acc = __builtin_amdgcn_wmma_f32_16x16x32_bf16(false, Ah, false, Blo[k], (short)0, acc, false, false);
            acc = __builtin_amdgcn_wmma_f32_16x16x32_bf16(false, Al, false, Bhi[k], (short)0, acc, false, false);
        }
        // acc[v] = <E[colBase + 8h + v], E[rowBase + r]>
        const float* xc = x2 + colBase + h * 8;
        float4 q0 = *(const float4*)(xc);
        float4 q1 = *(const float4*)(xc + 4);
        float xcv[8] = {q0.x, q0.y, q0.z, q0.w, q1.x, q1.y, q1.z, q1.w};
        const bool diagTile = (colBase == rowBase);   // wave-uniform
        #pragma unroll
        for (int v = 0; v < 8; ++v) {
            float d = fmaf(-2.0f, acc[v], xcv[v]);
            if (diagTile && (h * 8 + v) == r) d = BIGV;   // mask self
            if (d < m4) {                  // execz-skipped when no lane hits
                m4 = d;
                float t;
                t = fminf(m3, m4); m4 = fmaxf(m3, m4); m3 = t;
                t = fminf(m2, m3); m3 = fmaxf(m2, m3); m2 = t;
                t = fminf(m1, m2); m2 = fmaxf(m1, m2); m1 = t;
                t = fminf(m0, m1); m1 = fmaxf(m0, m1); m0 = t;
            }
        }
        stage_wait();       // my prefetch writes landed
        __syncthreads();    // everyone done writing buf^1 / reading buf
    }

    // --- Merge half-lane lists (bitonic split: A ++ rev(B) -> 5 smallest) ---
    float p0 = __shfl_xor(m0, 16, 32);
    float p1 = __shfl_xor(m1, 16, 32);
    float p2 = __shfl_xor(m2, 16, 32);
    float p3 = __shfl_xor(m3, 16, 32);
    float p4 = __shfl_xor(m4, 16, 32);
    float s0 = fminf(m0, p4), s1 = fminf(m1, p3), s2 = fminf(m2, p2),
          s3 = fminf(m3, p1), s4 = fminf(m4, p0);
    // Sort the 5 survivors (optimal 9-CE network) for the cross-wave merge.
    ce(s0, s1); ce(s3, s4); ce(s2, s4); ce(s2, s3);
    ce(s1, s4); ce(s0, s3); ce(s0, s2); ce(s1, s3); ce(s1, s2);

    if (lane < 16) {
        smem5[wave][r][0] = s0; smem5[wave][r][1] = s1; smem5[wave][r][2] = s2;
        smem5[wave][r][3] = s3; smem5[wave][r][4] = s4;
    }
    __syncthreads();

    // --- Cross-column-half merge + penalty (waves 0..3) ---
    if (wave < 4) {
        float q0 = smem5[wave + 4][r][0];
        float q1 = smem5[wave + 4][r][1];
        float q2 = smem5[wave + 4][r][2];
        float q3 = smem5[wave + 4][r][3];
        float q4 = smem5[wave + 4][r][4];
        float t0 = fminf(s0, q4), t1 = fminf(s1, q3), t2 = fminf(s2, q2),
              t3 = fminf(s3, q1), t4 = fminf(s4, q0);

        float mean = sqrtf(fmaxf(t0 + x2row, 1e-12f))
                   + sqrtf(fmaxf(t1 + x2row, 1e-12f))
                   + sqrtf(fmaxf(t2 + x2row, 1e-12f))
                   + sqrtf(fmaxf(t3 + x2row, 1e-12f))
                   + sqrtf(fmaxf(t4 + x2row, 1e-12f));
        mean *= 0.2f;
        float pen = (mean - 1.0f) * (mean - 1.0f);
        // Each row's penalty duplicated on lanes r and r+16 -> halve the sum.
        #pragma unroll
        for (int off = 16; off > 0; off >>= 1) pen += __shfl_xor(pen, off, 32);
        pen *= 0.5f;
        if (lane == 0) bsum[wave] = pen;
    }
    __syncthreads();
    if (threadIdx.x == 0)
        partials[blockIdx.x] = bsum[0] + bsum[1] + bsum[2] + bsum[3];
}

// ---------------------------------------------------------------------------
// Phase 3: deterministic fixed-order final reduction.
// ---------------------------------------------------------------------------
__global__ void finalize_kernel(const float* __restrict__ partials,
                                float* __restrict__ out) {
    if (threadIdx.x == 0 && blockIdx.x == 0) {
        float s = 0.0f;
        for (int i = 0; i < NBLK; ++i) s += partials[i];
        out[0] = s / (float)NPTS;
    }
}

extern "C" void kernel_launch(void* const* d_in, const int* in_sizes, int n_in,
                              void* d_out, int out_size, void* d_ws, size_t ws_size,
                              hipStream_t stream) {
    (void)in_sizes; (void)n_in; (void)out_size; (void)ws_size;
    const float* emb = (const float*)d_in[0];   // [16384,128] fp32; k==5 fixed

    char* ws = (char*)d_ws;
    unsigned short* ehi = (unsigned short*)ws;                      // 4 MB
    unsigned short* elo = ehi + (size_t)NPTS * DIM;                 // 4 MB
    float* x2       = (float*)(elo + (size_t)NPTS * DIM);           // 64 KB
    float* partials = x2 + NPTS;                                    // 1 KB

    prep_kernel<<<NPTS, DIM, 0, stream>>>(emb, ehi, elo, x2);
    knn_kernel<<<NBLK, WPB * 32, 0, stream>>>(ehi, elo, x2, partials);
    finalize_kernel<<<1, 32, 0, stream>>>(partials, (float*)d_out);
}
